// GGNNNet_57750130262285
// MI455X (gfx1250) — matrix operand order repaired and checked by reading, not compiled, not run on hardware
//
#include <hip/hip_runtime.h>
#include <hip/hip_bf16.h>
#include <math.h>

// ---------------------------------------------------------------------------
// Types for CDNA5 WMMA (wave32, 16x16x32 bf16 -> f32 accumulate)
// ---------------------------------------------------------------------------
typedef __bf16 v16bf __attribute__((ext_vector_type(16)));
typedef __bf16 v8bf  __attribute__((ext_vector_type(8)));
typedef float  v8f   __attribute__((ext_vector_type(8)));

static __device__ __forceinline__ v8f wmma_bf16(v16bf a, v16bf b, v8f c) {
    // (neg_a, A, neg_b, B, c_mod, C, reuse_a, reuse_b)
    return __builtin_amdgcn_wmma_f32_16x16x32_bf16(false, a, false, b,
                                                   (short)0, c, false, false);
}

static __device__ __forceinline__ v16bf cat8(v8bf lo, v8bf hi) {
    return __builtin_shufflevector(lo, hi, 0,1,2,3,4,5,6,7,8,9,10,11,12,13,14,15);
}

static __device__ __forceinline__ float softplusf(float x) {
    return x > 20.f ? x : log1pf(__expf(x));
}

#define N_NODES 100000
#define N_EDGES 1200000

// ---------------------------------------------------------------------------
// Small utility kernels
// ---------------------------------------------------------------------------
__global__ void k_cvt_bf16(const float* __restrict__ src, __bf16* __restrict__ dst, int n) {
    int i = blockIdx.x * blockDim.x + threadIdx.x;
    if (i < n) dst[i] = (__bf16)src[i];
}

__global__ void k_zero(float* __restrict__ p, int n) {
    int i = blockIdx.x * blockDim.x + threadIdx.x;
    if (i < n) p[i] = 0.f;
}

// per-edge softplus weight + degree accumulation (deg buffer reused as invdeg)
__global__ void k_edge_prep(const int* __restrict__ etype, const int* __restrict__ dst,
                            const float* __restrict__ etw,
                            float* __restrict__ we, float* __restrict__ deg, int E) {
    int e = blockIdx.x * blockDim.x + threadIdx.x;
    if (e >= E) return;
    we[e] = softplusf(etw[etype[e]]);
    atomicAdd(&deg[dst[e]], 1.0f);
}

__global__ void k_invdeg(float* __restrict__ deg, int n) {
    int i = blockIdx.x * blockDim.x + threadIdx.x;
    if (i < n) deg[i] = 1.0f / fmaxf(deg[i], 1.0f);
}

__global__ void k_alpha_init(float* __restrict__ alpha, int n) {
    int i = blockIdx.x * blockDim.x + threadIdx.x;
    if (i < n) alpha[i] = 1.0f;
}

__global__ void k_alpha_set(float* __restrict__ alpha, const int* __restrict__ diff,
                            const float* __restrict__ wimp, int n) {
    int i = blockIdx.x * blockDim.x + threadIdx.x;
    if (i < n) alpha[diff[i]] = 1.0f + softplusf(wimp[0]);   // idempotent (set, not add)
}

// ---------------------------------------------------------------------------
// Input MLP: h = relu(x @ W1^T + b1) @ W2^T + b2     (per 16-node tile)
// Layer1: [16,128]x[128,256]   Layer2: [16,256]x[256,64]
// ---------------------------------------------------------------------------
__global__ __launch_bounds__(256) void k_mlp(const float* __restrict__ x,
                                             const __bf16* __restrict__ w1,
                                             const float* __restrict__ b1,
                                             const __bf16* __restrict__ w2,
                                             const float* __restrict__ b2,
                                             float* __restrict__ h) {
    __shared__ __bf16 sA[16 * 128];    // x tile (bf16)
    __shared__ __bf16 sH1[16 * 256];   // relu(layer1) tile (bf16)
    const int tid = threadIdx.x;
    const int tileBase = blockIdx.x * 16;

    // stage + convert x tile (coalesced, 8 elems/thread)
    const float* xt = x + (size_t)tileBase * 128;
#pragma unroll
    for (int i = 0; i < 8; ++i) {
        int idx = tid + i * 256;
        sA[idx] = (__bf16)xt[idx];
    }
    __syncthreads();

    const int wave = tid >> 5, lane = tid & 31;
    const int ln = lane & 15, hi = lane >> 4;

    // ---- layer 1: 16 column tiles across 8 waves ----
    for (int t = wave; t < 16; t += 8) {
        const int nb = t * 16;
        v8f acc = {};
#pragma unroll
        for (int kb = 0; kb < 128; kb += 32) {
            const __bf16* pa = sA + ln * 128 + kb + hi * 16;
            v16bf afrag = cat8(*(const v8bf*)pa, *(const v8bf*)(pa + 8));
            v16bf bfrag = *(const v16bf*)(w1 + (nb + ln) * 128 + kb + hi * 16);
            acc = wmma_bf16(afrag, bfrag, acc);
        }
        const float bias = b1[nb + ln];
#pragma unroll
        for (int r = 0; r < 8; ++r) {
            int row = r + hi * 8;
            float v = acc[r] + bias;
            sH1[row * 256 + nb + ln] = (__bf16)(v > 0.f ? v : 0.f);
        }
    }
    __syncthreads();

    // ---- layer 2: 4 column tiles on waves 0..3 ----
    if (wave < 4) {
        const int nb = wave * 16;
        v8f acc = {};
#pragma unroll
        for (int kb = 0; kb < 256; kb += 32) {
            const __bf16* pa = sH1 + ln * 256 + kb + hi * 16;
            v16bf afrag = cat8(*(const v8bf*)pa, *(const v8bf*)(pa + 8));
            v16bf bfrag = *(const v16bf*)(w2 + (nb + ln) * 256 + kb + hi * 16);
            acc = wmma_bf16(afrag, bfrag, acc);
        }
        const float bias = b2[nb + ln];
#pragma unroll
        for (int r = 0; r < 8; ++r) {
            int row = r + hi * 8;
            h[(size_t)(tileBase + row) * 64 + nb + ln] = acc[r] + bias;
        }
    }
}

// ---------------------------------------------------------------------------
// Edge scatter: m[dst] += we * h[src]   (4 threads/edge, 16 channels each)
// ---------------------------------------------------------------------------
__global__ __launch_bounds__(256) void k_scatter(const float* __restrict__ h,
                                                 const int* __restrict__ src,
                                                 const int* __restrict__ dst,
                                                 const float* __restrict__ we,
                                                 float* __restrict__ m, int E) {
    int gid = blockIdx.x * 256 + threadIdx.x;
    int e = gid >> 2;
    if (e >= E) return;
    int q = gid & 3;
    int s = src[e], d = dst[e];
    float w = we[e];
    const float4* hs = (const float4*)(h + (size_t)s * 64 + q * 16);
    float* md = m + (size_t)d * 64 + q * 16;
#pragma unroll
    for (int j = 0; j < 4; ++j) {
        float4 v = hs[j];
        atomicAdd(md + j * 4 + 0, w * v.x);
        atomicAdd(md + j * 4 + 1, w * v.y);
        atomicAdd(md + j * 4 + 2, w * v.z);
        atomicAdd(md + j * 4 + 3, w * v.w);
    }
}

// ---------------------------------------------------------------------------
// Fused GRU cell per 16-node tile:
//   gi = (m*invdeg) @ Wih^T + bih ; gh = h @ Whh^T + bhh    (each [16,192])
//   r = sig(gi_r+gh_r); z = sig(gi_z+gh_z); n = tanh(gi_n + r*gh_n)
//   h = (1-z)*n + z*h        (in-place)
// ---------------------------------------------------------------------------
__global__ __launch_bounds__(256) void k_gru(float* __restrict__ h,
                                             const float* __restrict__ m,
                                             const float* __restrict__ invdeg,
                                             const __bf16* __restrict__ wih,
                                             const __bf16* __restrict__ whh,
                                             const float* __restrict__ bih,
                                             const float* __restrict__ bhh) {
    __shared__ __bf16 sM[16 * 64];
    __shared__ __bf16 sH[16 * 64];
    __shared__ float  sHold[16 * 64];
    __shared__ float  sPre[16 * 128];   // r,z pre-activations (gi+gh), cols 0..127
    __shared__ float  sGiN[16 * 64];    // gi_n
    __shared__ float  sGhN[16 * 64];    // gh_n
    const int tid = threadIdx.x;
    const int tileBase = blockIdx.x * 16;
    const size_t base = (size_t)tileBase * 64;

#pragma unroll
    for (int i = 0; i < 4; ++i) {
        int idx = tid + i * 256;
        int row = idx >> 6;
        float hv = h[base + idx];
        sHold[idx] = hv;
        sH[idx] = (__bf16)hv;
        sM[idx] = (__bf16)(m[base + idx] * invdeg[tileBase + row]);
    }
    __syncthreads();

    const int wave = tid >> 5, lane = tid & 31;
    const int ln = lane & 15, hi = lane >> 4;

    // 12 output column tiles (192 cols) over 8 waves
    for (int t = wave; t < 12; t += 8) {
        v8f ci = {};
        v8f ch = {};
#pragma unroll
        for (int kb = 0; kb < 64; kb += 32) {
            const __bf16* pam = sM + ln * 64 + kb + hi * 16;
            const __bf16* pah = sH + ln * 64 + kb + hi * 16;
            v16bf am = cat8(*(const v8bf*)pam, *(const v8bf*)(pam + 8));
            v16bf ah = cat8(*(const v8bf*)pah, *(const v8bf*)(pah + 8));
            v16bf bi = *(const v16bf*)(wih + (t * 16 + ln) * 64 + kb + hi * 16);
            v16bf bh = *(const v16bf*)(whh + (t * 16 + ln) * 64 + kb + hi * 16);
            ci = wmma_bf16(am, bi, ci);
            ch = wmma_bf16(ah, bh, ch);
        }
        const int col = t * 16 + ln;
        const float bi_ = bih[col], bh_ = bhh[col];
#pragma unroll
        for (int r = 0; r < 8; ++r) {
            int row = r + hi * 8;
            if (t < 8) {
                sPre[row * 128 + col] = ci[r] + ch[r] + bi_ + bh_;
            } else {
                sGiN[row * 64 + (col - 128)] = ci[r] + bi_;
                sGhN[row * 64 + (col - 128)] = ch[r] + bh_;
            }
        }
    }
    __syncthreads();

#pragma unroll
    for (int i = 0; i < 4; ++i) {
        int idx = tid + i * 256;
        int row = idx >> 6, c = idx & 63;
        float rg = 1.f / (1.f + __expf(-sPre[row * 128 + c]));
        float zg = 1.f / (1.f + __expf(-sPre[row * 128 + 64 + c]));
        float ng = tanhf(sGiN[idx] + rg * sGhN[idx]);
        h[base + idx] = (1.f - zg) * ng + zg * sHold[idx];
    }
}

// ---------------------------------------------------------------------------
// Importance-weighted pooling: pooled[c] = sum_n alpha[n]*h[n,c]; pooled[64] = sum alpha
// ---------------------------------------------------------------------------
__global__ __launch_bounds__(256) void k_pool(const float* __restrict__ h,
                                              const float* __restrict__ alpha,
                                              float* __restrict__ pooled, int N) {
    __shared__ float sRed[256];
    __shared__ float sAl[4];
    const int c = threadIdx.x & 63;
    const int rq = threadIdx.x >> 6;
    float acc = 0.f, aacc = 0.f;
    for (int node = blockIdx.x * 4 + rq; node < N; node += gridDim.x * 4) {
        float a = alpha[node];
        acc += a * h[(size_t)node * 64 + c];
        if (c == 0) aacc += a;
    }
    sRed[threadIdx.x] = acc;
    if (c == 0) sAl[rq] = aacc;
    __syncthreads();
    if (threadIdx.x < 64) {
        float s = sRed[threadIdx.x] + sRed[64 + threadIdx.x] +
                  sRed[128 + threadIdx.x] + sRed[192 + threadIdx.x];
        atomicAdd(&pooled[threadIdx.x], s);
    }
    if (threadIdx.x == 0) {
        atomicAdd(&pooled[64], sAl[0] + sAl[1] + sAl[2] + sAl[3]);
    }
}

// ---------------------------------------------------------------------------
// Head: out = log_softmax(relu(pooled @ fc1^T + b1) @ fc2^T + b2)
// ---------------------------------------------------------------------------
__global__ __launch_bounds__(256) void k_head(const float* __restrict__ pooled,
                                              const float* __restrict__ fc1w,
                                              const float* __restrict__ fc1b,
                                              const float* __restrict__ fc2w,
                                              const float* __restrict__ fc2b,
                                              float* __restrict__ out) {
    __shared__ float sP[64];
    __shared__ float sT[256];
    __shared__ float sO[2];
    const int tid = threadIdx.x;
    const float inv = 1.f / pooled[64];
    if (tid < 64) sP[tid] = pooled[tid] * inv;
    __syncthreads();
    float acc = fc1b[tid];
#pragma unroll 8
    for (int k = 0; k < 64; ++k) acc += sP[k] * fc1w[tid * 64 + k];
    sT[tid] = acc > 0.f ? acc : 0.f;
    __syncthreads();
    if (tid < 2) {
        float a = fc2b[tid];
        for (int k = 0; k < 256; ++k) a += sT[k] * fc2w[tid * 256 + k];
        sO[tid] = a;
    }
    __syncthreads();
    if (tid == 0) {
        float mx = fmaxf(sO[0], sO[1]);
        float l = mx + logf(__expf(sO[0] - mx) + __expf(sO[1] - mx));
        out[0] = sO[0] - l;
        out[1] = sO[1] - l;
    }
}

// ---------------------------------------------------------------------------
// Launcher
// ---------------------------------------------------------------------------
extern "C" void kernel_launch(void* const* d_in, const int* in_sizes, int n_in,
                              void* d_out, int out_size, void* d_ws, size_t ws_size,
                              hipStream_t stream) {
    const float* x    = (const float*)d_in[0];
    const int*   ei   = (const int*)d_in[1];     // [2,E] flat: src = ei, dst = ei+E
    const int*   ety  = (const int*)d_in[2];
    const int*   diff = (const int*)d_in[3];
    const float* l1w  = (const float*)d_in[4];
    const float* l1b  = (const float*)d_in[5];
    const float* l2w  = (const float*)d_in[6];
    const float* l2b  = (const float*)d_in[7];
    const float* gwih = (const float*)d_in[8];
    const float* gwhh = (const float*)d_in[9];
    const float* gbih = (const float*)d_in[10];
    const float* gbhh = (const float*)d_in[11];
    const float* etw  = (const float*)d_in[12];
    const float* f1w  = (const float*)d_in[13];
    const float* f1b  = (const float*)d_in[14];
    const float* f2w  = (const float*)d_in[15];
    const float* f2b  = (const float*)d_in[16];
    const float* wimp = (const float*)d_in[17];
    float* out = (float*)d_out;

    const int N = N_NODES, E = N_EDGES;
    const int nDiff = in_sizes[3];

    // workspace carve-out (256B aligned slices)
    char* ws = (char*)d_ws;
    size_t off = 0;
    auto carve = [&](size_t bytes) -> char* {
        char* p = ws + off;
        off += (bytes + 255) & ~(size_t)255;
        return p;
    };
    float*  h      = (float*)carve((size_t)N * 64 * 4);
    float*  mbuf   = (float*)carve((size_t)N * 64 * 4);
    float*  we     = (float*)carve((size_t)E * 4);
    float*  invdeg = (float*)carve((size_t)N * 4);   // deg, then 1/max(deg,1)
    float*  alpha  = (float*)carve((size_t)N * 4);
    float*  pooled = (float*)carve(65 * 4);
    __bf16* w1bf   = (__bf16*)carve(256 * 128 * 2);
    __bf16* w2bf   = (__bf16*)carve(64 * 256 * 2);
    __bf16* wihbf  = (__bf16*)carve(192 * 64 * 2);
    __bf16* whhbf  = (__bf16*)carve(192 * 64 * 2);

    // 1) weights -> bf16 (once per launch; tiny)
    k_cvt_bf16<<<(32768 + 255) / 256, 256, 0, stream>>>(l1w, w1bf, 32768);
    k_cvt_bf16<<<(16384 + 255) / 256, 256, 0, stream>>>(l2w, w2bf, 16384);
    k_cvt_bf16<<<(12288 + 255) / 256, 256, 0, stream>>>(gwih, wihbf, 12288);
    k_cvt_bf16<<<(12288 + 255) / 256, 256, 0, stream>>>(gwhh, whhbf, 12288);

    // 2) edge weights + degrees
    k_zero<<<(N + 255) / 256, 256, 0, stream>>>(invdeg, N);
    k_zero<<<1, 256, 0, stream>>>(pooled, 65);
    k_edge_prep<<<(E + 255) / 256, 256, 0, stream>>>(ety, ei + E, etw, we, invdeg, E);
    k_invdeg<<<(N + 255) / 256, 256, 0, stream>>>(invdeg, N);

    // 3) input MLP (WMMA)
    k_mlp<<<N / 16, 256, 0, stream>>>(x, w1bf, l1b, w2bf, l2b, h);

    // 4) 3 rounds of message passing + fused GRU (WMMA)
    for (int it = 0; it < 3; ++it) {
        k_zero<<<(N * 64 + 255) / 256, 256, 0, stream>>>(mbuf, N * 64);
        k_scatter<<<((E * 4) + 255) / 256, 256, 0, stream>>>(h, ei, ei + E, we, mbuf, E);
        k_gru<<<N / 16, 256, 0, stream>>>(h, mbuf, invdeg, wihbf, whhbf, gbih, gbhh);
    }

    // 5) pooling + head
    k_alpha_init<<<(N + 255) / 256, 256, 0, stream>>>(alpha, N);
    k_alpha_set<<<(nDiff + 255) / 256, 256, 0, stream>>>(alpha, diff, wimp, nDiff);
    k_pool<<<512, 256, 0, stream>>>(h, alpha, pooled, N);
    k_head<<<1, 256, 0, stream>>>(pooled, f1w, f1b, f2w, f2b, out);
}
